// SlidingWindowAttention_7773890805828
// MI455X (gfx1250) — compile-verified, matrix operand order checked by reading, not run on hardware
//
#include <hip/hip_runtime.h>
#include <hip/hip_bf16.h>

#define BB 8
#define SS 2048
#define DD 1024
#define AA 512

typedef float v8f __attribute__((ext_vector_type(8)));
typedef __bf16 v16bf __attribute__((ext_vector_type(16)));

__device__ inline unsigned short f2bf(float f) {
    __bf16 h = (__bf16)f;                          // native v_cvt_*_bf16_f32
    return __builtin_bit_cast(unsigned short, h);
}
__device__ inline float bf2f(unsigned short h) {
    unsigned u = ((unsigned)h) << 16;
    return __builtin_bit_cast(float, u);
}

struct U16x8 { unsigned short v[8]; };

__device__ inline U16x8 load8(const unsigned short* p) {
    uint4 q = *(const uint4*)p;                    // 8 bf16 in one b128
    return __builtin_bit_cast(U16x8, q);
}
__device__ inline U16x8 load8(const float* p) {
    float4 lo = *(const float4*)p;                 // 2 x b128 + native cvt
    float4 hi = *(const float4*)(p + 4);
    U16x8 r;
    r.v[0] = f2bf(lo.x); r.v[1] = f2bf(lo.y); r.v[2] = f2bf(lo.z); r.v[3] = f2bf(lo.w);
    r.v[4] = f2bf(hi.x); r.v[5] = f2bf(hi.y); r.v[6] = f2bf(hi.z); r.v[7] = f2bf(hi.w);
    return r;
}

// CDNA5 async copy: 16 bytes global -> LDS per lane, tracked by ASYNCcnt.
__device__ inline void async_copy_b128(void* lds, const void* g) {
    unsigned loff = (unsigned)(unsigned long long)(size_t)lds;  // low 32b = LDS offset
    unsigned long long ga = (unsigned long long)(size_t)g;
    asm volatile("global_load_async_to_lds_b128 %0, %1, off"
                 :: "v"(loff), "v"(ga) : "memory");
}
__device__ inline void wait_async() {
    asm volatile("s_wait_asynccnt 0x0" ::: "memory");
}

union FragBF { uint4 q[2]; v16bf v; };

// ---------------------------------------------------------------------------
// Tiled WMMA GEMM: C[m,n] (+epilogue) = sum_k A[m,k] * B(k,n)
//   Block tile 128x128, 8 waves (256 threads), wave tile 32x64 = 2x4 WMMA.
//   Double-buffered LDS, K-step 32, bf16 operands, f32 accumulate.
//   bf16 operand tiles are staged with GLOBAL_LOAD_ASYNC_TO_LDS_B128.
//   EPI: 0 = bf16 out, acc + bias[n]              (QKV projection)
//        1 = f32  out, acc * alpha                (attention scores)
//        2 = bf16 out, alpha*acc + beta*extra[n]  (weights@v + global blend)
//        3 = f32  out, acc + bias[n] + extra[m,n] (out-proj + residual)
// ---------------------------------------------------------------------------
template <typename AT, typename BT, bool B_N_CONTIG, int EPI>
__global__ __launch_bounds__(256) void gemm_wmma(
    const AT* __restrict__ Ag, long lda, long bsA,
    const BT* __restrict__ Bg, long ldbk, long ldbn, long bsB,
    void* __restrict__ Cg, long ldc, long bsC,
    const float* __restrict__ bias,
    const float* __restrict__ extra, long bsE,
    float alpha, float beta, int K)
{
    __shared__ __align__(16) unsigned short ldsA[2][128 * 40];
    __shared__ __align__(16) unsigned short ldsB[2][128 * 40];

    constexpr bool ASYNC_A = __is_same(AT, unsigned short);
    constexpr bool ASYNC_B = (!B_N_CONTIG) && __is_same(BT, unsigned short);

    const int tid   = threadIdx.x;
    const int lane  = tid & 31;
    const int wave  = tid >> 5;
    const int waveM = (wave >> 1) * 32;   // 4 waves along M
    const int waveN = (wave & 1) * 64;    // 2 waves along N
    const int bM    = blockIdx.x * 128;
    const int bN    = blockIdx.y * 128;
    const int z     = blockIdx.z;
    const int mrow  = lane & 15;
    const int half  = lane >> 4;          // K-half select (ISA 16-bit frag layout)

    const AT* Ab = Ag + (long)z * bsA;
    const BT* Bb = Bg + (long)z * bsB;

    // 128x32 A tile: 4096 elems, 8-elem (16B) chunks -> 2 chunks/thread
    auto fillA = [&](int buf, int k0) {
        #pragma unroll
        for (int it = 0; it < 2; ++it) {
            int c   = it * 256 + tid;
            int r   = c >> 2;
            int col = (c & 3) * 8;
            const AT* src      = Ab + (long)(bM + r) * lda + (k0 + col);
            unsigned short* dst = &ldsA[buf][r * 40 + col];
            if constexpr (ASYNC_A) {
                async_copy_b128(dst, src);
            } else {
                U16x8 d = load8(src);
                *(uint4*)dst = __builtin_bit_cast(uint4, d);
            }
        }
    };
    // 128x32 B tile stored [n][k] in LDS
    auto fillB = [&](int buf, int k0) {
        #pragma unroll
        for (int it = 0; it < 2; ++it) {
            int c = it * 256 + tid;
            if constexpr (B_N_CONTIG) {
                // memory [K,N], n contiguous: vector load 8 n, scalar transpose into LDS
                int kk = c >> 4;
                int n8 = (c & 15) * 8;
                U16x8 d = load8(Bb + (long)(k0 + kk) * ldbk + (long)(bN + n8));
                #pragma unroll
                for (int e = 0; e < 8; ++e)
                    ldsB[buf][(n8 + e) * 40 + kk] = d.v[e];
            } else {
                // memory [N,K], k contiguous: straight 16B copy
                int n   = c >> 2;
                int col = (c & 3) * 8;
                const BT* src      = Bb + (long)(bN + n) * ldbn + (k0 + col);
                unsigned short* dst = &ldsB[buf][n * 40 + col];
                if constexpr (ASYNC_B) {
                    async_copy_b128(dst, src);
                } else {
                    U16x8 d = load8(src);
                    *(uint4*)dst = __builtin_bit_cast(uint4, d);
                }
            }
        }
    };

    v8f acc[2][4];
    #pragma unroll
    for (int i = 0; i < 2; ++i)
        #pragma unroll
        for (int j = 0; j < 4; ++j)
            #pragma unroll
            for (int e = 0; e < 8; ++e) acc[i][j][e] = 0.0f;

    fillA(0, 0);
    fillB(0, 0);
    int cur = 0;
    for (int k0 = 0; k0 < K; k0 += 32) {
        if constexpr (ASYNC_A || ASYNC_B) wait_async();  // ASYNCcnt -> 0 before barrier
        __syncthreads();
        if (k0 + 32 < K) {              // stage next tile into the other buffer
            fillA(cur ^ 1, k0 + 32);
            fillB(cur ^ 1, k0 + 32);
        }
        if (k0 + 64 < K) {              // gfx1250 prefetch of the tile after that
            __builtin_prefetch(Ab + (long)(bM + (tid >> 1)) * lda + (k0 + 64), 0, 1);
        }

        FragBF a[2], b[4];
        #pragma unroll
        for (int t = 0; t < 2; ++t) {
            const unsigned short* pa = &ldsA[cur][(waveM + t * 16 + mrow) * 40 + half * 8];
            a[t].q[0] = *(const uint4*)pa;         // K = half*8 + 0..7
            a[t].q[1] = *(const uint4*)(pa + 16);  // K = 16 + half*8 + 0..7
        }
        #pragma unroll
        for (int t = 0; t < 4; ++t) {
            const unsigned short* pb = &ldsB[cur][(waveN + t * 16 + mrow) * 40 + half * 8];
            b[t].q[0] = *(const uint4*)pb;
            b[t].q[1] = *(const uint4*)(pb + 16);
        }
        #pragma unroll
        for (int i = 0; i < 2; ++i)
            #pragma unroll
            for (int j = 0; j < 4; ++j)
                acc[i][j] = __builtin_amdgcn_wmma_f32_16x16x32_bf16(
                    false, a[i].v, false, b[j].v, (short)0, acc[i][j], false, false);
        cur ^= 1;
    }

    #pragma unroll
    for (int i = 0; i < 2; ++i) {
        #pragma unroll
        for (int j = 0; j < 4; ++j) {
            #pragma unroll
            for (int r = 0; r < 8; ++r) {
                int m = bM + waveM + i * 16 + r + 8 * half;  // C layout: VGPR r / lane half
                int n = bN + waveN + j * 16 + mrow;
                float v = acc[i][j][r];
                long off = (long)z * bsC + (long)m * ldc + n;
                if constexpr (EPI == 0) {
                    ((unsigned short*)Cg)[off] = f2bf(v + bias[n]);
                } else if constexpr (EPI == 1) {
                    ((float*)Cg)[off] = v * alpha;
                } else if constexpr (EPI == 2) {
                    ((unsigned short*)Cg)[off] =
                        f2bf(alpha * v + beta * extra[(long)z * bsE + n]);
                } else {
                    ((float*)Cg)[off] = v + bias[n] + extra[(long)z * bsE + (long)m * ldc + n];
                }
            }
        }
    }
}

// --------------------------- scalar support kernels -------------------------

__global__ __launch_bounds__(256) void gq_kernel(
    const float* __restrict__ gc, const float* __restrict__ Wq,
    const float* __restrict__ bq, float* __restrict__ gq)
{
    int b = blockIdx.y;
    int a = blockIdx.x * 256 + threadIdx.x;
    const float* x = gc + (long)b * DD;
    float s = bq[a];
    for (int d = 0; d < DD; ++d) s += x[d] * Wq[(long)d * AA + a];
    gq[b * AA + a] = s;
}

__global__ __launch_bounds__(256) void gsoftmax_kernel(
    const float* __restrict__ gq, const unsigned short* __restrict__ kbf,
    float* __restrict__ gw)
{
    __shared__ float sc[SS];
    __shared__ float red[256];
    int b = blockIdx.x, tid = threadIdx.x;
    const float* q = gq + (long)b * AA;
    const unsigned short* kb = kbf + (long)b * SS * AA;
    const float inv = 0.044194173824159216f;  // 1/sqrt(512)
    for (int j = tid; j < SS; j += 256) {
        float s = 0.f;
        const unsigned short* kr = kb + (long)j * AA;
        for (int a = 0; a < AA; ++a) s += q[a] * bf2f(kr[a]);
        sc[j] = s * inv;
    }
    __syncthreads();
    float m = -1e30f;
    for (int j = tid; j < SS; j += 256) m = fmaxf(m, sc[j]);
    red[tid] = m; __syncthreads();
    for (int s = 128; s > 0; s >>= 1) { if (tid < s) red[tid] = fmaxf(red[tid], red[tid + s]); __syncthreads(); }
    m = red[0]; __syncthreads();
    float sum = 0.f;
    for (int j = tid; j < SS; j += 256) { float e = __expf(sc[j] - m); sc[j] = e; sum += e; }
    red[tid] = sum; __syncthreads();
    for (int s = 128; s > 0; s >>= 1) { if (tid < s) red[tid] += red[tid + s]; __syncthreads(); }
    float is = 1.0f / red[0];
    for (int j = tid; j < SS; j += 256) gw[(long)b * SS + j] = sc[j] * is;
}

__global__ __launch_bounds__(256) void gout_kernel(
    const float* __restrict__ gw, const unsigned short* __restrict__ vbf,
    float* __restrict__ gout)
{
    int b = blockIdx.y;
    int a = blockIdx.x * 256 + threadIdx.x;
    const float* w = gw + (long)b * SS;
    const unsigned short* vb = vbf + (long)b * SS * AA;
    float s = 0.f;
    for (int j = 0; j < SS; ++j) s += w[j] * bf2f(vb[(long)j * AA + a]);
    gout[b * AA + a] = s;
}

__global__ __launch_bounds__(256) void softmax_kernel(
    const float* __restrict__ scores, unsigned short* __restrict__ wout)
{
    __shared__ float row[SS];
    __shared__ float red[256];
    long r = blockIdx.x;  // B*S rows
    int tid = threadIdx.x;
    const float* src = scores + r * SS;
    float m = -1e30f;
    for (int j = tid; j < SS; j += 256) { float v = src[j]; row[j] = v; m = fmaxf(m, v); }
    red[tid] = m; __syncthreads();
    for (int s = 128; s > 0; s >>= 1) { if (tid < s) red[tid] = fmaxf(red[tid], red[tid + s]); __syncthreads(); }
    m = red[0]; __syncthreads();
    float sum = 0.f;
    for (int j = tid; j < SS; j += 256) { float e = __expf(row[j] - m); row[j] = e; sum += e; }
    red[tid] = sum; __syncthreads();
    for (int s = 128; s > 0; s >>= 1) { if (tid < s) red[tid] += red[tid + s]; __syncthreads(); }
    float is = 1.0f / red[0];
    unsigned short* dst = wout + r * SS;
    for (int j = tid; j < SS; j += 256) dst[j] = f2bf(row[j] * is);
}

__global__ __launch_bounds__(256) void avgw_kernel(
    const unsigned short* __restrict__ w, const float* __restrict__ gw,
    float* __restrict__ avg_out)
{
    int b = blockIdx.y;
    int j = blockIdx.x * 256 + threadIdx.x;
    const unsigned short* wb = w + (long)b * SS * SS + j;
    float s = 0.f;
    for (int q = 0; q < SS; ++q) s += bf2f(wb[(long)q * SS]);
    avg_out[(long)b * SS + j] = 0.7f * (s / (float)SS) + 0.3f * gw[(long)b * SS + j];
}

__global__ __launch_bounds__(256) void layernorm_kernel(
    const float* __restrict__ y, const float* __restrict__ gamma,
    const float* __restrict__ beta, float* __restrict__ yln)
{
    __shared__ float red[256], red2[256];
    long r = blockIdx.x;  // B*S rows
    int tid = threadIdx.x;
    const float* src = y + r * DD;
    float s = 0.f, s2 = 0.f;
    for (int d = tid; d < DD; d += 256) { float v = src[d]; s += v; s2 += v * v; }
    red[tid] = s; red2[tid] = s2; __syncthreads();
    for (int k = 128; k > 0; k >>= 1) {
        if (tid < k) { red[tid] += red[tid + k]; red2[tid] += red2[tid + k]; }
        __syncthreads();
    }
    float mu = red[0] / (float)DD;
    float var = red2[0] / (float)DD - mu * mu;
    float rstd = rsqrtf(var + 1e-5f);
    float* dst = yln + r * DD;
    for (int d = tid; d < DD; d += 256) dst[d] = (src[d] - mu) * rstd * gamma[d] + beta[d];
}

__global__ __launch_bounds__(256) void context_kernel(
    const float* __restrict__ avg, const float* __restrict__ yln,
    float* __restrict__ ctx)
{
    int b = blockIdx.y;
    int d = blockIdx.x * 256 + threadIdx.x;
    const float* a = avg + (long)b * SS;
    const float* y = yln + (long)b * SS * DD + d;
    float s = 0.f;
    for (int j = 0; j < SS; ++j) s += a[j] * y[(long)j * DD];
    ctx[(long)b * DD + d] = s;
}

// ---------------------------------------------------------------------------

extern "C" void kernel_launch(void* const* d_in, const int* in_sizes, int n_in,
                              void* d_out, int out_size, void* d_ws, size_t ws_size,
                              hipStream_t stream)
{
    const float* x     = (const float*)d_in[0];   // [B,S,D]
    const float* gc    = (const float*)d_in[1];   // [B,1,D]
    const float* Wq    = (const float*)d_in[2];   // [D,A]
    const float* bq    = (const float*)d_in[3];
    const float* Wk    = (const float*)d_in[4];
    const float* bk    = (const float*)d_in[5];
    const float* Wv    = (const float*)d_in[6];
    const float* bv    = (const float*)d_in[7];
    const float* Wo    = (const float*)d_in[8];   // [A,D]
    const float* bo    = (const float*)d_in[9];
    const float* gamma = (const float*)d_in[10];
    const float* beta  = (const float*)d_in[11];

    const size_t MB = 1ull << 20;
    char* ws = (char*)d_ws;
    unsigned short* qb   = (unsigned short*)(ws);              // 16 MB bf16 q
    unsigned short* kb   = (unsigned short*)(ws + 16 * MB);    // 16 MB bf16 k
    unsigned short* vb   = (unsigned short*)(ws + 32 * MB);    // 16 MB bf16 v
    float*          sc   = (float*)(ws + 48 * MB);             // 128 MB f32 scores
    float*          y    = (float*)(ws + 48 * MB);             // aliases scores (dead)
    float*          yln  = (float*)(ws + 112 * MB);            // aliases scores top half
    unsigned short* wb   = (unsigned short*)(ws + 176 * MB);   // 64 MB bf16 weights
    unsigned short* wout = (unsigned short*)(ws + 240 * MB);   // 16 MB bf16 win_out
    float*          gq   = (float*)(ws + 256 * MB);            // 16 KB
    float*          gw   = (float*)(ws + 256 * MB + 64 * 1024);   // 64 KB
    float*          gob  = (float*)(ws + 256 * MB + 192 * 1024);  // 16 KB

    float* avgw = (float*)d_out + (long)BB * DD;  // [B,S]
    float* ctx  = (float*)d_out;                  // [B,D]

    const dim3 blk(256);
    const float inv_scale = 0.044194173824159216f;  // 1/sqrt(A)

    // 1) QKV projections: [B*S,D] x [D,A] -> bf16
    {
        dim3 g((BB * SS) / 128, AA / 128, 1);
        gemm_wmma<float, float, true, 0><<<g, blk, 0, stream>>>(
            x, DD, 0, Wq, AA, 1, 0, qb, AA, 0, bq, nullptr, 0, 1.f, 0.f, DD);
        gemm_wmma<float, float, true, 0><<<g, blk, 0, stream>>>(
            x, DD, 0, Wk, AA, 1, 0, kb, AA, 0, bk, nullptr, 0, 1.f, 0.f, DD);
        gemm_wmma<float, float, true, 0><<<g, blk, 0, stream>>>(
            x, DD, 0, Wv, AA, 1, 0, vb, AA, 0, bv, nullptr, 0, 1.f, 0.f, DD);
    }

    // 2) global-context query projection
    gq_kernel<<<dim3(AA / 256, BB), 256, 0, stream>>>(gc, Wq, bq, gq);

    // 3) scores = q @ k^T / sqrt(A)  -> f32 [B,S,S]  (async-to-LDS staging)
    gemm_wmma<unsigned short, unsigned short, false, 1>
        <<<dim3(SS / 128, SS / 128, BB), blk, 0, stream>>>(
            qb, AA, (long)SS * AA, kb, 1, AA, (long)SS * AA,
            sc, SS, (long)SS * SS, nullptr, nullptr, 0, inv_scale, 0.f, AA);

    // 4) row softmax -> bf16 weights
    softmax_kernel<<<BB * SS, 256, 0, stream>>>(sc, wb);

    // 5) global scores + softmax, 6) global_out = gweights @ v
    gsoftmax_kernel<<<BB, 256, 0, stream>>>(gq, kb, gw);
    gout_kernel<<<dim3(AA / 256, BB), 256, 0, stream>>>(gw, vb, gob);

    // 7) avg_w = mean_q(0.7*weights) + 0.3*gweights  -> d_out tail
    avgw_kernel<<<dim3(SS / 256, BB), 256, 0, stream>>>(wb, gw, avgw);

    // 8) win_out = 0.7*(weights @ v) + 0.3*global_out  -> bf16
    gemm_wmma<unsigned short, unsigned short, true, 2>
        <<<dim3(SS / 128, AA / 128, BB), blk, 0, stream>>>(
            wb, SS, (long)SS * SS, vb, AA, 1, (long)SS * AA,
            wout, AA, (long)SS * AA, nullptr, gob, AA, 0.7f, 0.3f, SS);

    // 9) y = x + win_out @ Wo + bo  -> f32 (scores region is dead, reuse it)
    gemm_wmma<unsigned short, float, true, 3>
        <<<dim3((BB * SS) / 128, DD / 128, 1), blk, 0, stream>>>(
            wout, AA, 0, Wo, DD, 1, 0, y, DD, 0, bo, x, 0, 1.f, 0.f, AA);

    // 10) layernorm rows
    layernorm_kernel<<<BB * SS, 256, 0, stream>>>(y, gamma, beta, yln);

    // 11) context = avg_w-weighted sum of y_ln rows -> d_out head
    context_kernel<<<dim3(DD / 256, BB), 256, 0, stream>>>(avgw, yln, ctx);
}
